// Attention_25168508354564
// MI455X (gfx1250) — compile-verified, matrix operand order checked by reading, not run on hardware
//
#include <hip/hip_runtime.h>
#include <hip/hip_bf16.h>

typedef __attribute__((ext_vector_type(16))) __bf16 v16bf;
typedef __attribute__((ext_vector_type(8)))  float  v8f;
typedef __attribute__((ext_vector_type(4)))  float  v4f;
typedef __attribute__((ext_vector_type(2)))  unsigned u2;

#define NH 16
#define T_SEQ 2048
#define DH 64
#define DMODEL 1024

static __device__ inline __bf16 f2bf(float f) {
  unsigned u = __float_as_uint(f);
  u += 0x7FFFu + ((u >> 16) & 1u);          // round-to-nearest-even
  unsigned short s = (unsigned short)(u >> 16);
  return __builtin_bit_cast(__bf16, s);
}
static __device__ inline unsigned pack2bf(float a, float b) {
  unsigned ua = __float_as_uint(a); ua += 0x7FFFu + ((ua >> 16) & 1u);
  unsigned ub = __float_as_uint(b); ub += 0x7FFFu + ((ub >> 16) & 1u);
  return (ua >> 16) | (ub & 0xFFFF0000u);
}

// ---- WMMA fragment loaders (CDNA5 wave32 layouts, 16x16x32 bf16) ----
static __device__ inline v16bf load_a_frag(const __bf16* A, int row0, int k0, int ld) {
  int lane = threadIdx.x & 31;
  int m = lane & 15;
  int kb = (lane < 16) ? 0 : 8;
  const __bf16* p = A + (size_t)(row0 + m) * ld + k0 + kb;
  v16bf a;
#pragma unroll
  for (int i = 0; i < 8; ++i) { a[i] = p[i]; a[i + 8] = p[16 + i]; }
  return a;
}
static __device__ inline v16bf load_b_frag(const __bf16* B, int k0, int col0, int ld) {
  int lane = threadIdx.x & 31;
  int n = lane & 15;
  int kb = (lane < 16) ? 0 : 8;
  const __bf16* p = B + (size_t)(k0 + kb) * ld + col0 + n;
  v16bf b;
#pragma unroll
  for (int i = 0; i < 8; ++i) { b[i] = p[i * ld]; b[i + 8] = p[(16 + i) * ld]; }
  return b;
}
static __device__ inline v16bf load_kT_frag(const __bf16* Kg, size_t bh, int key0, int kOff) {
  int lane = threadIdx.x & 31;
  int n = lane & 15;
  int kb = (lane < 16) ? 0 : 8;
  const __bf16* p = Kg + (bh * T_SEQ + key0 + n) * DH + kOff + kb;
  v16bf v;
#pragma unroll
  for (int i = 0; i < 8; ++i) { v[i] = p[i]; v[i + 8] = p[16 + i]; }
  return v;
}
static __device__ inline v16bf load_v_frag(const __bf16* Vg, size_t bh, int kc, int n0) {
  int lane = threadIdx.x & 31;
  int nn = lane & 15;
  int kb = (lane < 16) ? 0 : 8;
  const __bf16* p = Vg + (bh * T_SEQ + kc + kb) * DH + n0 + nn;
  v16bf v;
#pragma unroll
  for (int i = 0; i < 8; ++i) { v[i] = p[i * DH]; v[i + 8] = p[(16 + i) * DH]; }
  return v;
}

// ---------------- f32 -> bf16 conversion ----------------
__global__ void cvt_f32_bf16(const float* __restrict__ in, __bf16* __restrict__ out, int n) {
  int i = blockIdx.x * blockDim.x + threadIdx.x;
  if (i < n) out[i] = f2bf(in[i]);
}

// ---------------- tiled bf16 WMMA GEMM ----------------
#define BM 128
#define BN 64
#define BK 32

template <int MODE>
__global__ __launch_bounds__(256) void gemm_bf16_kernel(
    const __bf16* __restrict__ A, const __bf16* __restrict__ B,
    int M, int N, int Kdim, float* __restrict__ Cout,
    __bf16* __restrict__ Qb, __bf16* __restrict__ Kb, __bf16* __restrict__ Vb) {
  __shared__ __bf16 As[BM * BK];
  __shared__ __bf16 Bs[BK * BN];
  const int t = threadIdx.x;
  const int lane = t & 31;
  const int w = t >> 5;
  const int wm = w >> 1;
  const int wn = w & 1;
  const int m0 = blockIdx.y * BM;
  const int n0 = blockIdx.x * BN;

  v8f acc[2][2] = {};
  for (int k0 = 0; k0 < Kdim; k0 += BK) {
    {
      int r = t >> 1, cb = (t & 1) * 16;
      const __bf16* src = A + (size_t)(m0 + r) * Kdim + k0 + cb;
      __bf16* dst = As + r * BK + cb;
      *(uint4*)(dst) = *(const uint4*)(src);
      *(uint4*)(dst + 8) = *(const uint4*)(src + 8);
    }
    {
      int r = t >> 3, cb = (t & 7) * 8;
      *(uint4*)(Bs + r * BN + cb) = *(const uint4*)(B + (size_t)(k0 + r) * N + n0 + cb);
    }
    __syncthreads();
#pragma unroll
    for (int mi = 0; mi < 2; ++mi) {
      v16bf a = load_a_frag(As, wm * 32 + mi * 16, 0, BK);
#pragma unroll
      for (int ni = 0; ni < 2; ++ni) {
        v16bf bf = load_b_frag(Bs, 0, wn * 32 + ni * 16, BN);
        acc[mi][ni] = __builtin_amdgcn_wmma_f32_16x16x32_bf16(
            false, a, false, bf, (short)0, acc[mi][ni], false, false);
      }
    }
    __syncthreads();
  }

  const int nlane = lane & 15;
  const int mbase = (lane < 16) ? 0 : 8;
#pragma unroll
  for (int mi = 0; mi < 2; ++mi)
#pragma unroll
    for (int ni = 0; ni < 2; ++ni) {
      int rb = m0 + wm * 32 + mi * 16;
      int c = n0 + wn * 32 + ni * 16 + nlane;
#pragma unroll
      for (int i = 0; i < 8; ++i) {
        int r = rb + mbase + i;
        float v = acc[mi][ni][i];
        if (MODE == 0) {
          Cout[(size_t)r * N + c] = v;
        } else {
          int bb = r >> 11, tt = r & (T_SEQ - 1);
          int which = c >> 10, rem = c & (DMODEL - 1);
          int h = rem >> 6, d = rem & (DH - 1);
          __bf16* dst = (which == 0) ? Qb : (which == 1) ? Kb : Vb;
          dst[(((size_t)bb * NH + h) * T_SEQ + tt) * DH + d] = f2bf(v);
        }
      }
    }
}

// ---------------- fused masked-softmax attention ----------------
__global__ __launch_bounds__(256) void attn_kernel(
    const __bf16* __restrict__ Qg, const __bf16* __restrict__ Kg,
    const __bf16* __restrict__ Vg, const float* __restrict__ prior,
    const unsigned char* __restrict__ qmask, float* __restrict__ attn_score,
    float* __restrict__ attn_prob, __bf16* __restrict__ Yb) {
  extern __shared__ char smem[];
  float* S = (float*)smem;                                          // 16*2048 f32  (128KB)
  __bf16* P = (__bf16*)(smem + 16 * T_SEQ * 4);                     // 16*2048 bf16 (64KB)
  __bf16* Qs = (__bf16*)(smem + 16 * T_SEQ * 4 + 16 * T_SEQ * 2);   // 16*64 bf16   (2KB)
  unsigned char* kmS = (unsigned char*)(smem + 16 * T_SEQ * 4 + 16 * T_SEQ * 2 + 16 * DH * 2); // 2048B
  float* red = (float*)(smem + 16 * T_SEQ * 4 + 16 * T_SEQ * 2 + 16 * DH * 2 + T_SEQ);         // 1KB

  const int t = threadIdx.x;
  const int lane = t & 31;
  const int w = t >> 5;
  const int qb = blockIdx.x & 127;
  const int h = (blockIdx.x >> 7) & (NH - 1);
  const int b = blockIdx.x >> 11;
  const int q0 = qb << 4;
  const size_t bh = (size_t)b * NH + h;
  const float scale = 0.125f;  // 1/sqrt(64)
  const float NEG_INF = -__builtin_inff();

  {  // stage Q tile (16x64 bf16) and the batch's mask row (2048B) into LDS
    int i = t * 4;
    int r = i >> 6, c = i & (DH - 1);
    *(uint2*)(Qs + i) = *(const uint2*)(Qg + (bh * T_SEQ + q0 + r) * DH + c);
    *(uint2*)(kmS + t * 8) = *(const uint2*)(qmask + (size_t)b * T_SEQ + t * 8);
  }
  __syncthreads();

  const int nlane = lane & 15;
  const int mbase = (lane < 16) ? 0 : 8;

  // row-valid (pad mask) bits for this lane's 8 output rows -- hoisted out of key loop
  bool mrow[8];
#pragma unroll
  for (int i = 0; i < 8; ++i) mrow[i] = kmS[q0 + mbase + i] != 0;

  // ---- pass 1: S = scale * Q K^T, causal+pad mask, stream attn_score ----
  v16bf aq0 = load_a_frag(Qs, 0, 0, DH);
  v16bf aq1 = load_a_frag(Qs, 0, 32, DH);
  for (int j = w; j < T_SEQ / 16; j += 8) {
    int k0 = j << 4;
    int kg = k0 + nlane;
    if (j <= qb) {
      v8f acc = {};
      v16bf b0 = load_kT_frag(Kg, bh, k0, 0);
      acc = __builtin_amdgcn_wmma_f32_16x16x32_bf16(false, aq0, false, b0, (short)0, acc, false, false);
      v16bf b1 = load_kT_frag(Kg, bh, k0, 32);
      acc = __builtin_amdgcn_wmma_f32_16x16x32_bf16(false, aq1, false, b1, (short)0, acc, false, false);
      bool mk = kmS[kg] != 0;
#pragma unroll
      for (int i = 0; i < 8; ++i) {
        int r = mbase + i;
        int qg = q0 + r;
        bool ok = (kg <= qg) & mk & mrow[i];
        float v = ok ? acc[i] * scale : NEG_INF;       // branchless v_cndmask
        S[r * T_SEQ + kg] = v;
        __builtin_nontemporal_store(v, &attn_score[(bh * T_SEQ + qg) * T_SEQ + kg]);
      }
    } else {  // fully above the causal diagonal
#pragma unroll
      for (int i = 0; i < 8; ++i) {
        int r = mbase + i;
        int qg = q0 + r;
        S[r * T_SEQ + kg] = NEG_INF;
        __builtin_nontemporal_store(NEG_INF, &attn_score[(bh * T_SEQ + qg) * T_SEQ + kg]);
      }
    }
  }
  __syncthreads();

  // ---- pass 2: softmax w/ prior reweighting: p ~ exp(s-max)*(prior+tiny) ----
  const int r2 = t >> 4, seg = t & 15;
  float* Sr = S + r2 * T_SEQ;
  const int c0 = seg * (T_SEQ / 16);     // 128 cols per thread, 16B-aligned
  v4f* Sr4 = (v4f*)(Sr + c0);

  float m = NEG_INF;
  for (int c = 0; c < 32; ++c) {
    v4f s4 = Sr4[c];
    m = fmaxf(m, fmaxf(fmaxf(s4.x, s4.y), fmaxf(s4.z, s4.w)));
  }
  red[r2 * 16 + seg] = m;
  __syncthreads();
  float rowmax = NEG_INF;
#pragma unroll
  for (int i = 0; i < 16; ++i) rowmax = fmaxf(rowmax, red[r2 * 16 + i]);
  __syncthreads();

  const float tiny = 1.17549435e-38f;    // finfo(f32).tiny
  const int qg2 = q0 + r2;
  const v4f* pr4 = (const v4f*)(prior + ((size_t)b * T_SEQ + qg2) * T_SEQ + c0);
  float sum = 0.f;
  for (int c = 0; c < 32; ++c) {
    v4f s4 = Sr4[c];
    v4f p4 = pr4[c];
    v4f w4;
    w4.x = expf(s4.x - rowmax) * (p4.x + tiny);
    w4.y = expf(s4.y - rowmax) * (p4.y + tiny);
    w4.z = expf(s4.z - rowmax) * (p4.z + tiny);
    w4.w = expf(s4.w - rowmax) * (p4.w + tiny);
    Sr4[c] = w4;
    sum += (w4.x + w4.y) + (w4.z + w4.w);
  }
  red[r2 * 16 + seg] = sum;
  __syncthreads();
  float rowsum = 0.f;
#pragma unroll
  for (int i = 0; i < 16; ++i) rowsum += red[r2 * 16 + i];
  float inv = 1.0f / rowsum;

  v4f* prob4 = (v4f*)(attn_prob + (bh * T_SEQ + qg2) * T_SEQ + c0);
  u2* P2 = (u2*)(P + r2 * T_SEQ + c0);
  for (int c = 0; c < 32; ++c) {
    v4f w4 = Sr4[c];
    v4f o4 = w4 * inv;
    __builtin_nontemporal_store(o4, &prob4[c]);
    u2 pk;
    pk.x = pack2bf(o4.x, o4.y);
    pk.y = pack2bf(o4.z, o4.w);
    P2[c] = pk;
  }
  __syncthreads();

  // ---- pass 3: y = P V (waves 0..3 own one 16-wide slice of dh) ----
  if (w < 4) {
    int n0 = w * 16;
    v8f acc = {};
    int kend = q0 + 16;   // P is exactly 0 beyond the causal diagonal
    for (int kc = 0; kc < kend; kc += 32) {
      v16bf pa = load_a_frag(P, 0, kc, T_SEQ);
      v16bf vb = load_v_frag(Vg, bh, kc, n0);
      acc = __builtin_amdgcn_wmma_f32_16x16x32_bf16(false, pa, false, vb, (short)0, acc, false, false);
    }
#pragma unroll
    for (int i = 0; i < 8; ++i) {
      int r = mbase + i;
      int qg = q0 + r;
      Yb[((size_t)b * T_SEQ + qg) * DMODEL + h * DH + n0 + nlane] = f2bf(acc[i]);
    }
  }
}

extern "C" void kernel_launch(void* const* d_in, const int* in_sizes, int n_in,
                              void* d_out, int out_size, void* d_ws, size_t ws_size,
                              hipStream_t stream) {
  const float* query = (const float*)d_in[0];
  const unsigned char* qmask = (const unsigned char*)d_in[1];
  const float* prior = (const float*)d_in[2];
  const float* Wqkv = (const float*)d_in[3];
  const float* Wo = (const float*)d_in[4];

  float* out = (float*)d_out;
  float* attn_prob = out + (size_t)4194304;
  float* attn_score = attn_prob + (size_t)134217728;

  char* ws = (char*)d_ws;
  __bf16* Xbf = (__bf16*)ws;      ws += (size_t)4096 * 1024 * 2;
  __bf16* Wqkvbf = (__bf16*)ws;   ws += (size_t)1024 * 3072 * 2;
  __bf16* Wobf = (__bf16*)ws;     ws += (size_t)1024 * 1024 * 2;
  __bf16* Qb = (__bf16*)ws;       ws += (size_t)2 * NH * T_SEQ * DH * 2;
  __bf16* Kb = (__bf16*)ws;       ws += (size_t)2 * NH * T_SEQ * DH * 2;
  __bf16* Vb = (__bf16*)ws;       ws += (size_t)2 * NH * T_SEQ * DH * 2;
  __bf16* Yb = (__bf16*)ws;       ws += (size_t)4096 * 1024 * 2;

  cvt_f32_bf16<<<(4096 * 1024 + 255) / 256, 256, 0, stream>>>(query, Xbf, 4096 * 1024);
  cvt_f32_bf16<<<(1024 * 3072 + 255) / 256, 256, 0, stream>>>(Wqkv, Wqkvbf, 1024 * 3072);
  cvt_f32_bf16<<<(1024 * 1024 + 255) / 256, 256, 0, stream>>>(Wo, Wobf, 1024 * 1024);

  dim3 g1(3072 / BN, 4096 / BM);
  gemm_bf16_kernel<1><<<g1, 256, 0, stream>>>(Xbf, Wqkvbf, 4096, 3072, 1024,
                                              nullptr, Qb, Kb, Vb);

  size_t lds = (size_t)16 * T_SEQ * 4 + (size_t)16 * T_SEQ * 2 + 16 * DH * 2 + T_SEQ + 16 * 16 * 4;
  attn_kernel<<<2 * NH * (T_SEQ / 16), 256, lds, stream>>>(
      Qb, Kb, Vb, prior, qmask, attn_score, attn_prob, Yb);

  dim3 g2(1024 / BN, 4096 / BM);
  gemm_bf16_kernel<0><<<g2, 256, 0, stream>>>(Yb, Wobf, 4096, 1024, 1024,
                                              out, nullptr, nullptr, nullptr);
}
